// Shift_Module_68564857913585
// MI455X (gfx1250) — matrix operand order repaired and checked by reading, not compiled, not run on hardware
//
#include <hip/hip_runtime.h>
#include <hip/hip_bf16.h>

typedef __attribute__((ext_vector_type(16))) __bf16 v16bf;
typedef __attribute__((ext_vector_type(8)))  float  v8f;

#define C_CH   256
#define Hdim   96
#define Wdim   96
#define HW     (Hdim*Wdim)          // 9216
#define KTOT   (C_CH*9)             // 2304
#define KCHUNKS (KTOT/32)           // 72
#define BATCH  2
#define NPOS   (BATCH*HW)           // 18432
#define NTILES (HW/16)              // 576 per image
#define PADW   98
#define PADHW  (PADW*PADW)          // 9604 (96x96 with 1-px zero halo)

// workspace layout (bytes)
#define METAW_OFF 0u
#define METAI_OFF 2654208u          // NPOS*9*16
#define PACKA_OFF 5308416u          // w_conv bf16 packed: 1,179,648 B
#define PACKO_OFF 6488064u          // w_off  bf16 packed:    147,456 B
#define BNFLD_OFF 6561792u          // folded BN:              2,048 B
#define XPAD_OFF  6563840u          // padded x: 19,668,992 B -> total ~26.3 MB

__device__ __forceinline__ int clampi(int v, int lo, int hi) {
    return v < lo ? lo : (v > hi ? hi : v);
}

// A-operand K index for (lane, halfword): contiguous K order, K = 16*(lane>=16) + hw
__device__ __forceinline__ int a_layout_K(int lane, int hw) {
    return hw + ((lane < 16) ? 0 : 16);
}

// ---------------------------------------------------------------------------
// Zero-padded copy of x: xpad[b][c][98][98] (1-px halo). One-time stream.
// ---------------------------------------------------------------------------
__global__ __launch_bounds__(256)
void pad_x_kernel(const float* __restrict__ x, float* __restrict__ xpad)
{
    int id  = blockIdx.x * 256 + threadIdx.x;     // exact: B*C*9604 = 256*19208
    int bc  = id / PADHW;
    int rc  = id - bc * PADHW;
    int row = rc / PADW;
    int col = rc - row * PADW;
    int h = row - 1, w = col - 1;
    bool ok = ((unsigned)h < (unsigned)Hdim) & ((unsigned)w < (unsigned)Wdim);
    int hc = clampi(h, 0, Hdim - 1), wc = clampi(w, 0, Wdim - 1);
    float v = x[(size_t)bc * HW + hc * Wdim + wc];
    xpad[id] = ok ? v : 0.0f;
}

// ---------------------------------------------------------------------------
// Pack w_conv (256x2304 fp32) -> bf16, pre-permuted into per-lane A layout.
// ---------------------------------------------------------------------------
__global__ __launch_bounds__(256)
void pack_wconv_kernel(const float* __restrict__ wconv, __bf16* __restrict__ packA)
{
    int wid   = blockIdx.x * 256 + threadIdx.x;   // u32 word id, 0..294911
    int g     = wid >> 8;
    int r     = wid & 255;
    int mtile = g / KCHUNKS;
    int kc    = g - mtile * KCHUNKS;
    int lane  = r >> 3;
    int j     = r & 7;
    int o     = mtile * 16 + (lane & 15);
    #pragma unroll
    for (int half = 0; half < 2; ++half) {
        int K = a_layout_K(lane, 2 * j + half);
        packA[(size_t)wid * 2 + half] = (__bf16)wconv[(size_t)o * KTOT + kc * 32 + K];
    }
}

// ---------------------------------------------------------------------------
// Pack w_off (27x2304 fp32) -> bf16 A layout, zero-padded to 32 rows.
// ---------------------------------------------------------------------------
__global__ __launch_bounds__(256)
void pack_woff_kernel(const float* __restrict__ woff, __bf16* __restrict__ packO)
{
    int wid   = blockIdx.x * 256 + threadIdx.x;   // 0..36863
    int g     = wid >> 8;
    int r     = wid & 255;
    int mtile = g / KCHUNKS;                      // 0..1
    int kc    = g - mtile * KCHUNKS;
    int lane  = r >> 3;
    int j     = r & 7;
    int o     = mtile * 16 + (lane & 15);
    #pragma unroll
    for (int half = 0; half < 2; ++half) {
        int K = a_layout_K(lane, 2 * j + half);
        float f = (o < 27) ? woff[(size_t)o * KTOT + kc * 32 + K] : 0.0f;
        packO[(size_t)wid * 2 + half] = (__bf16)f;
    }
}

// ---------------------------------------------------------------------------
// Fold BN+bias into (scale, shift) per output channel. One block of 256.
// ---------------------------------------------------------------------------
__global__ __launch_bounds__(256)
void bn_fold_kernel(const float* __restrict__ b_conv, const float* __restrict__ gamma,
                    const float* __restrict__ beta,   const float* __restrict__ rmean,
                    const float* __restrict__ rvar,   float2* __restrict__ bn)
{
    int o = threadIdx.x;
    float sc = gamma[o] * rsqrtf(rvar[o] + 1e-5f);
    float sh = beta[o] + (b_conv[o] - rmean[o]) * sc;
    bn[o] = make_float2(sc, sh);
}

// ---------------------------------------------------------------------------
// Kernel 1: offset conv as WMMA GEMM (M=32 padded, N=128/block, K=2304);
// im2col from padded x with purely incremental 32-bit addressing:
// K += 32 = 3 channels + 1 row + 2 cols  =>  addr += 3*9604 + 98 + 2,
// +95 on column wrap, +9310 on row wrap. No div/clamp/64-bit math per element.
// ---------------------------------------------------------------------------
__global__ __launch_bounds__(256)
void offset_meta_kernel(const float* __restrict__ xpad,
                        const __bf16* __restrict__ packO,
                        const float* __restrict__ b_off,
                        float4* __restrict__ metaW,
                        int4*  __restrict__ metaI)
{
    __shared__ __attribute__((aligned(32))) __bf16 Bbuf[2][8 * 32 * 16]; // 2 x 8KB
    __shared__ float accS[8][32][16];                                    // 16 KB

    const int tid     = threadIdx.x;
    const int lane    = tid & 31;
    const int wave    = tid >> 5;
    const int pos0    = blockIdx.x * 128;        // 9216 % 128 == 0: no b straddle
    const int b       = pos0 / HW;
    const int hw_base = pos0 - b * HW;
    const float* xpb  = xpad + (size_t)b * C_CH * PADHW;

    // thread stages column n = tid&127, K rows i = (tid>>7) + 2r, r = 0..15
    const int ncol  = tid & 127;
    const int ibase = tid >> 7;
    const int hwn   = hw_base + ncol;
    const int hpos  = hwn / Wdim;
    const int wpos  = hwn - hpos * Wdim;
    const int grp   = ncol >> 4;
    const int nn    = ncol & 15;
    const int base0 = hpos * PADW + wpos;        // tap (dh,dw) -> +dh*98+dw in padded coords

    int adA[16], dhA[16], dwA[16];
    #pragma unroll
    for (int r = 0; r < 16; ++r) {               // one-time init divisions
        int i   = ibase + 2 * r;
        int c   = i / 9;
        int tap = i - 9 * c;
        int dh  = tap / 3;
        int dw  = tap - 3 * dh;
        dhA[r]  = dh;
        dwA[r]  = dw;
        adA[r]  = c * PADHW + dh * PADW + dw + base0;
    }

    auto stage = [&](int buf) {                   // must be called in kc order
        #pragma unroll
        for (int r = 0; r < 16; ++r) {
            int i = ibase + 2 * r;
            float v = xpb[adA[r]];                // always-legal (halo-padded)
            int blane = (i < 16) ? nn : (nn + 16);
            Bbuf[buf][(grp * 32 + blane) * 16 + (i & 15)] = (__bf16)v;
            // advance K by 32 taps
            int dw2 = dwA[r] + 2;  bool cw = dw2 >= 3;  dw2 = cw ? dw2 - 3 : dw2;
            int dh2 = dhA[r] + (cw ? 2 : 1);
            bool ch = dh2 >= 3;    dh2 = ch ? dh2 - 3 : dh2;
            adA[r] += 3 * PADHW + PADW + 2 + (cw ? (PADW - 3) : 0)
                                            + (ch ? (PADHW - 3 * PADW) : 0);
            dwA[r] = dw2;
            dhA[r] = dh2;
        }
    };

    v8f acc0 = {};
    v8f acc1 = {};
    stage(0);
    for (int kc = 0; kc < KCHUNKS; ++kc) {
        __syncthreads();                          // staging of buf[kc&1] visible
        v16bf bv = *(const v16bf*)&Bbuf[kc & 1][(wave * 32 + lane) * 16];
        if (kc + 1 < KCHUNKS) stage((kc + 1) & 1);
        v16bf av0 = *(const v16bf*)(packO + ((size_t)(0 * KCHUNKS + kc) * 32 + lane) * 16);
        acc0 = __builtin_amdgcn_wmma_f32_16x16x32_bf16(false, av0, false, bv,
                                                       (short)0, acc0, false, false);
        v16bf av1 = *(const v16bf*)(packO + ((size_t)(1 * KCHUNKS + kc) * 32 + lane) * 16);
        acc1 = __builtin_amdgcn_wmma_f32_16x16x32_bf16(false, av1, false, bv,
                                                       (short)0, acc1, false, false);
    }

    // spill D tiles to LDS: row M = v + 8*(lane>=16), col = lane&15
    const int col = lane & 15;
    const int rhi = (lane >> 4) * 8;
    __syncthreads();
    #pragma unroll
    for (int v = 0; v < 8; ++v) {
        accS[wave][rhi + v][col]      = acc0[v];
        accS[wave][16 + rhi + v][col] = acc1[v];
    }
    __syncthreads();

    // 128 threads compute bilinear metadata (9 taps each)
    if (tid < 128) {
        const int n  = tid;
        const int g  = n >> 4;
        const int cc = n & 15;
        const int hw = hw_base + n;
        const int h  = hw / Wdim;
        const int w  = hw - h * Wdim;
        float om[27];
        #pragma unroll
        for (int j = 0; j < 27; ++j) om[j] = accS[g][j][cc] + b_off[j];

        #pragma unroll
        for (int k = 0; k < 9; ++k) {
            float dy = om[2 * k], dx = om[2 * k + 1];
            float m  = 1.0f / (1.0f + __expf(-om[18 + k]));
            float py = (float)h + (float)(k / 3) - 1.0f + dy;
            float px = (float)w + (float)(k % 3) - 1.0f + dx;
            float y0f = floorf(py), x0f = floorf(px);
            float ly = py - y0f, lx = px - x0f;
            int y0 = (int)y0f, x0 = (int)x0f;
            int y1 = y0 + 1,   x1 = x0 + 1;
            float vy0 = (y0 >= 0 && y0 < Hdim) ? 1.0f : 0.0f;
            float vy1 = (y1 >= 0 && y1 < Hdim) ? 1.0f : 0.0f;
            float vx0 = (x0 >= 0 && x0 < Wdim) ? 1.0f : 0.0f;
            float vx1 = (x1 >= 0 && x1 < Wdim) ? 1.0f : 0.0f;
            float4 wv;
            wv.x = (1.0f - ly) * (1.0f - lx) * m * vy0 * vx0;
            wv.y = (1.0f - ly) * lx          * m * vy0 * vx1;
            wv.z = ly          * (1.0f - lx) * m * vy1 * vx0;
            wv.w = ly          * lx          * m * vy1 * vx1;
            int cy0 = clampi(y0, 0, Hdim - 1), cy1 = clampi(y1, 0, Hdim - 1);
            int cx0 = clampi(x0, 0, Wdim - 1), cx1 = clampi(x1, 0, Wdim - 1);
            int4 iv;
            iv.x = cy0 * Wdim + cx0;  iv.y = cy0 * Wdim + cx1;
            iv.z = cy1 * Wdim + cx0;  iv.w = cy1 * Wdim + cx1;
            metaW[(size_t)(b * 9 + k) * HW + hw] = wv;
            metaI[(size_t)(b * 9 + k) * HW + hw] = iv;
        }
    }
}

// ---------------------------------------------------------------------------
// Kernel 2: fused bilinear-gather + GEMM (M=256 x N=16 per block), B tile
// double-buffered in LDS; 32-bit incremental channel-base addressing.
// ---------------------------------------------------------------------------
__global__ __launch_bounds__(256)
void deform_gemm_kernel(const float*  __restrict__ x,
                        const float4* __restrict__ metaW,
                        const int4*   __restrict__ metaI,
                        const __bf16* __restrict__ packA,
                        const float2* __restrict__ bn,
                        float* __restrict__ out)
{
    __shared__ float4 mwS[9 * 16];
    __shared__ int4   miS[9 * 16];
    __shared__ float2 bnS[C_CH];
    __shared__ __attribute__((aligned(32))) __bf16 Bbuf[2][32 * 16];

    const int tid    = threadIdx.x;
    const int bt     = blockIdx.x;
    const int b      = bt / NTILES;
    const int tile   = bt - b * NTILES;
    const int n_base = tile * 16;
    const float* xb  = x + (size_t)b * C_CH * HW;

    bnS[tid] = bn[tid];                           // 256 channels, 256 threads
    for (int idx = tid; idx < 144; idx += 256) {
        int k = idx >> 4, n = idx & 15;
        mwS[idx] = metaW[(size_t)(b * 9 + k) * HW + n_base + n];
        miS[idx] = metaI[(size_t)(b * 9 + k) * HW + n_base + n];
    }

    const int lane = tid & 31;
    const int wave = tid >> 5;
    const int n    = tid & 15;

    // this thread stages K rows i0 = tid>>4 (0..15) and i0+16, column n
    const int i0 = tid >> 4;
    int cB[2], kA[2];
    #pragma unroll
    for (int rpt = 0; rpt < 2; ++rpt) {
        int i   = i0 + rpt * 16;
        int c   = i / 9;
        kA[rpt] = i - 9 * c;
        cB[rpt] = c * HW;                         // 32-bit channel base offset
    }

    auto stage = [&](int buf) {                   // called in kc order
        #pragma unroll
        for (int rpt = 0; rpt < 2; ++rpt) {
            int i = i0 + rpt * 16;
            int k = kA[rpt];
            int cb = cB[rpt];
            float4 wv = mwS[k * 16 + n];
            int4   iv = miS[k * 16 + n];
            float s = wv.x * xb[cb + iv.x] + wv.y * xb[cb + iv.y]
                    + wv.z * xb[cb + iv.z] + wv.w * xb[cb + iv.w];
            int blane = (i < 16) ? n : (n + 16);
            Bbuf[buf][blane * 16 + (i & 15)] = (__bf16)s;
            int t2 = k + 5;                       // K += 32 = 3*9+5
            bool wrap = t2 >= 9;
            kA[rpt] = wrap ? t2 - 9 : t2;
            cB[rpt] = cb + (wrap ? 4 * HW : 3 * HW);
        }
    };

    __syncthreads();                              // meta/bn staged
    v8f acc0 = {};
    v8f acc1 = {};
    stage(0);
    for (int kc = 0; kc < KCHUNKS; ++kc) {
        __syncthreads();                          // buf[kc&1] ready; prior reads done
        v16bf bv = *(const v16bf*)&Bbuf[kc & 1][lane * 16];
        if (kc + 1 < KCHUNKS) stage((kc + 1) & 1);
        v16bf av0 = *(const v16bf*)(packA + ((size_t)(wave * KCHUNKS + kc) * 32 + lane) * 16);
        acc0 = __builtin_amdgcn_wmma_f32_16x16x32_bf16(false, av0, false, bv,
                                                       (short)0, acc0, false, false);
        v16bf av1 = *(const v16bf*)(packA + ((size_t)((wave + 8) * KCHUNKS + kc) * 32 + lane) * 16);
        acc1 = __builtin_amdgcn_wmma_f32_16x16x32_bf16(false, av1, false, bv,
                                                       (short)0, acc1, false, false);
    }

    // epilogue: D layout — VGPR v, lane: M = v + 8*(lane>=16), N = lane&15
    const int pos   = n_base + n;
    const int rowHi = (lane >> 4) * 8;
    #pragma unroll
    for (int t = 0; t < 2; ++t) {
        int obase = ((t == 0) ? wave : (wave + 8)) * 16;
        v8f a = (t == 0) ? acc0 : acc1;
        #pragma unroll
        for (int v = 0; v < 8; ++v) {
            int o = obase + rowHi + v;
            float2 p = bnS[o];
            float y  = fmaxf(a[v] * p.x + p.y, 0.0f);
            size_t oidx = ((size_t)b * C_CH + o) * HW + pos;
            out[oidx] = x[oidx] + y;
        }
    }
}

// ---------------------------------------------------------------------------
extern "C" void kernel_launch(void* const* d_in, const int* in_sizes, int n_in,
                              void* d_out, int out_size, void* d_ws, size_t ws_size,
                              hipStream_t stream)
{
    const float* x      = (const float*)d_in[0];
    const float* w_off  = (const float*)d_in[1];
    const float* b_off  = (const float*)d_in[2];
    const float* w_conv = (const float*)d_in[3];
    const float* b_conv = (const float*)d_in[4];
    const float* gamma  = (const float*)d_in[5];
    const float* beta   = (const float*)d_in[6];
    const float* rmean  = (const float*)d_in[7];
    const float* rvar   = (const float*)d_in[8];
    float* out = (float*)d_out;

    char* ws = (char*)d_ws;
    float4* metaW = (float4*)(ws + METAW_OFF);
    int4*   metaI = (int4*)(ws + METAI_OFF);
    __bf16* packA = (__bf16*)(ws + PACKA_OFF);
    __bf16* packO = (__bf16*)(ws + PACKO_OFF);
    float2* bn    = (float2*)(ws + BNFLD_OFF);
    float*  xpad  = (float*)(ws + XPAD_OFF);     // total ws use ~26.3 MB

    pad_x_kernel<<<(BATCH * C_CH * PADHW) / 256, 256, 0, stream>>>(x, xpad);
    pack_wconv_kernel<<<1152, 256, 0, stream>>>(w_conv, packA);
    pack_woff_kernel<<<144, 256, 0, stream>>>(w_off, packO);
    bn_fold_kernel<<<1, 256, 0, stream>>>(b_conv, gamma, beta, rmean, rvar, bn);
    offset_meta_kernel<<<NPOS / 128, 256, 0, stream>>>(xpad, packO, b_off, metaW, metaI);
    deform_gemm_kernel<<<BATCH * NTILES, 256, 0, stream>>>(x, metaW, metaI, packA, bn, out);
}
